// SelfAttention_82205674045992
// MI455X (gfx1250) — matrix operand order repaired and checked by reading, not compiled, not run on hardware
//
#include <hip/hip_runtime.h>
#include <hip/hip_bf16.h>

// ---------------- types ----------------
typedef __attribute__((ext_vector_type(16))) __bf16        v16bf;
typedef __attribute__((ext_vector_type(8)))  float         v8f;
typedef __attribute__((ext_vector_type(4)))  unsigned int  u32x4;
typedef __attribute__((ext_vector_type(8)))  int           i32x8;
typedef __attribute__((ext_vector_type(4)))  int           i32x4;

#define HIDDEN 256
#define LSEQ   4096
#define BATCH  4
#define KV_T   32            // keys per kv tile
#define NTILES (LSEQ / KV_T) // 128

__device__ __forceinline__ v8f wmma_bf16(v16bf a, v16bf b, v8f c) {
  // D = A(16x32 bf16) * B(32x16 bf16) + C(16x16 f32)
  return __builtin_amdgcn_wmma_f32_16x16x32_bf16(
      /*neg_a=*/false, a, /*neg_b=*/false, b,
      /*c_mod=*/(short)0, c, /*reuse_a=*/false, /*reuse_b=*/false);
}

__device__ __forceinline__ unsigned lds_off(const void* p) {
  // generic LDS pointer: addr[31:0] is the LDS byte offset
  return (unsigned)(unsigned long long)p;
}

// 2D tile load Global->LDS via Tensor Data Mover (bf16 elements).
// tile_d0 = contiguous elements per row, tile_d1 = rows, stride = row pitch (elems).
__device__ __forceinline__ void tdm_load_2d_bf16(unsigned ldsaddr, const void* gptr,
                                                 unsigned tile_d0, unsigned tile_d1,
                                                 unsigned stride) {
  unsigned long long ga = (unsigned long long)gptr;
  u32x4 g0;
  g0[0] = 1u;                                   // count=1, user descriptor
  g0[1] = ldsaddr;                              // lds_addr
  g0[2] = (unsigned)(ga & 0xFFFFFFFFu);         // global_addr[31:0]
  g0[3] = (unsigned)((ga >> 32) & 0x1FFFFFFu)   // global_addr[56:32]
        | (2u << 30);                           // type = 2 ("image")
  unsigned td0 = stride;                        // tensor_dim0 = full row extent
  unsigned td1 = tile_d1;                       // tensor_dim1
  i32x8 g1;
  g1[0] = (int)(1u << 16);                      // data_size = 1 (2 bytes)
  g1[1] = (int)((td0 & 0xFFFFu) << 16);                             // tensor_dim0[15:0]
  g1[2] = (int)(((td0 >> 16) & 0xFFFFu) | ((td1 & 0xFFFFu) << 16)); // td0[31:16] | td1[15:0]
  g1[3] = (int)(((td1 >> 16) & 0xFFFFu) | ((tile_d0 & 0xFFFFu) << 16)); // td1[31:16] | tile_dim0
  g1[4] = (int)(tile_d1 & 0xFFFFu);             // tile_dim1 | tile_dim2=0
  g1[5] = (int)stride;                          // tensor_dim0_stride[31:0]
  g1[6] = 0;
  g1[7] = 0;
  i32x4 gz4 = {0, 0, 0, 0};                     // 2D tensor: groups 2/3 unused
  i32x8 gz8 = {0, 0, 0, 0, 0, 0, 0, 0};
  // 6-arg form (this toolchain): (g0, g1, g2, g3, extra, cpol)
  __builtin_amdgcn_tensor_load_to_lds(g0, g1, gz4, gz4, gz8, 0);
}

// ================= Phase 1: QKV projection =================
// C[M=16384, N=256] = X[16384x256] * W[256x256]; block = 8 waves x 16 rows,
// 64 N-cols per block; z selects Wq/Wk/Wv. Outputs bf16 (V transposed).
__global__ __launch_bounds__(256)
void qkv_gemm_kernel(const float* __restrict__ X,
                     const float* __restrict__ Wq,
                     const float* __restrict__ Wk,
                     const float* __restrict__ Wv,
                     __bf16* __restrict__ Qb,
                     __bf16* __restrict__ Kb,
                     __bf16* __restrict__ Vt) {
  const int lane = threadIdx.x & 31;
  const int wid  = threadIdx.x >> 5;
  const int hi   = lane >> 4;      // half-wave select
  const int ll   = lane & 15;
  const int m0   = blockIdx.x * 128 + wid * 16;  // row base of this wave
  const int n0   = blockIdx.y * 64;              // col base of this block
  const int z    = blockIdx.z;
  const float* __restrict__ W = (z == 0) ? Wq : (z == 1) ? Wk : Wv;

  v8f acc[4] = {};

  for (int k0 = 0; k0 < HIDDEN; k0 += 32) {
    // A fragment: 16x32 bf16. lane row = m0+ll; elem j -> K = k0+16*hi+j.
    const float* ap = X + (size_t)(m0 + ll) * HIDDEN + k0 + 16 * hi;
    v16bf a;
#pragma unroll
    for (int j = 0; j < 8; ++j) { a[j] = (__bf16)ap[j]; a[8 + j] = (__bf16)ap[8 + j]; }
#pragma unroll
    for (int c = 0; c < 4; ++c) {
      // B fragment: 32x16 bf16. lane col = n0+c*16+ll; elem j -> K = k0+16*hi+j.
      const int n = n0 + c * 16 + ll;
      v16bf b;
#pragma unroll
      for (int j = 0; j < 16; ++j)
        b[j] = (__bf16)W[(size_t)(k0 + 16 * hi + j) * HIDDEN + n];
      acc[c] = wmma_bf16(a, b, acc[c]);
    }
  }

#pragma unroll
  for (int c = 0; c < 4; ++c) {
#pragma unroll
    for (int i = 0; i < 8; ++i) {
      const int row = m0 + i + 8 * hi;            // global M index (b*4096 + l)
      const int col = n0 + c * 16 + ll;
      const __bf16 v = (__bf16)acc[c][i];
      if (z == 0)      Qb[(size_t)row * HIDDEN + col] = v;
      else if (z == 1) Kb[(size_t)row * HIDDEN + col] = v;
      else {
        const int bb = row >> 12, lrow = row & (LSEQ - 1);
        Vt[((size_t)bb * HIDDEN + col) * LSEQ + lrow] = v;  // V transposed
      }
    }
  }
}

// ================= Phase 2: flash attention =================
// block = (64 queries, 1 batch); 4 waves x 16 q-rows. K/V tiles of 32 keys
// staged in LDS by TDM, double-buffered. Online softmax in fp32.
__global__ __launch_bounds__(128)
void attn_kernel(const __bf16* __restrict__ Qb,
                 const __bf16* __restrict__ Kb,
                 const __bf16* __restrict__ Vt,
                 float* __restrict__ out) {
  __shared__ __align__(64) __bf16 sK[2][KV_T][HIDDEN];   // 2*16KB
  __shared__ __align__(64) __bf16 sV[2][HIDDEN][KV_T];   // 2*16KB (V^T tile)
  __shared__ __align__(64) __bf16 sP[4][16][KV_T];       // 4KB, per-wave P

  const int lane = threadIdx.x & 31;
  const int wid  = threadIdx.x >> 5;
  const int hi   = lane >> 4;
  const int ll   = lane & 15;
  const int b    = blockIdx.y;
  const int q0   = blockIdx.x * 64 + wid * 16;

  const __bf16* __restrict__ Qrow  = Qb + ((size_t)b * LSEQ + q0 + ll) * HIDDEN;
  const __bf16* __restrict__ Kbase = Kb + (size_t)b * LSEQ * HIDDEN;
  const __bf16* __restrict__ Vbase = Vt + (size_t)b * HIDDEN * LSEQ;

  v8f O[16] = {};
  v8f m, l;
#pragma unroll
  for (int i = 0; i < 8; ++i) { m[i] = -1e30f; l[i] = 0.f; }

  if (wid == 0) {  // prologue: DMA tile 0 into buffer 0
    tdm_load_2d_bf16(lds_off(&sK[0][0][0]), Kbase, HIDDEN, KV_T, HIDDEN);
    tdm_load_2d_bf16(lds_off(&sV[0][0][0]), Vbase, KV_T, HIDDEN, LSEQ);
  }

  for (int t = 0; t < NTILES; ++t) {
    const int buf = t & 1;
    if (wid == 0) {
      if (t + 1 < NTILES) {  // prefetch next tile into other buffer
        tdm_load_2d_bf16(lds_off(&sK[buf ^ 1][0][0]),
                         Kbase + (size_t)(t + 1) * KV_T * HIDDEN, HIDDEN, KV_T, HIDDEN);
        tdm_load_2d_bf16(lds_off(&sV[buf ^ 1][0][0]),
                         Vbase + (size_t)(t + 1) * KV_T, KV_T, HIDDEN, LSEQ);
        __builtin_amdgcn_s_wait_tensorcnt(2);  // current tile's 2 DMAs done
      } else {
        __builtin_amdgcn_s_wait_tensorcnt(0);
      }
    }
    __syncthreads();  // publish LDS tile to all waves

    // ---- S = Q * K^T (16 x 32), fp32 accum ----
    v8f S0 = {}, S1 = {};
#pragma unroll
    for (int kd = 0; kd < 8; ++kd) {
      const int d0 = kd * 32;
      const __bf16* ap = Qrow + d0 + 16 * hi;
      v16bf a;
#pragma unroll
      for (int j = 0; j < 8; ++j) { a[j] = ap[j]; a[8 + j] = ap[8 + j]; }
      // B-frag: elem j -> dim d0 + 16*hi + j, key = ll (S0) / 16+ll (S1)
      v16bf k0f = *(const v16bf*)&sK[buf][ll][d0 + 16 * hi];
      v16bf k1f = *(const v16bf*)&sK[buf][16 + ll][d0 + 16 * hi];
      S0 = wmma_bf16(a, k0f, S0);
      S1 = wmma_bf16(a, k1f, S1);
    }

    // ---- online softmax (rows live in 16-lane halves) ----
    v8f r;
#pragma unroll
    for (int i = 0; i < 8; ++i) {
      S0[i] *= 0.0625f;  // 1/sqrt(256)
      S1[i] *= 0.0625f;
      r[i] = fmaxf(S0[i], S1[i]);
    }
#pragma unroll
    for (int msk = 1; msk < 16; msk <<= 1)
#pragma unroll
      for (int i = 0; i < 8; ++i) r[i] = fmaxf(r[i], __shfl_xor(r[i], msk, 32));

    v8f alpha, ps;
#pragma unroll
    for (int i = 0; i < 8; ++i) {
      const float mn = fmaxf(m[i], r[i]);
      alpha[i] = __expf(m[i] - mn);
      m[i] = mn;
      S0[i] = __expf(S0[i] - mn);
      S1[i] = __expf(S1[i] - mn);
      ps[i] = S0[i] + S1[i];
    }
#pragma unroll
    for (int msk = 1; msk < 16; msk <<= 1)
#pragma unroll
      for (int i = 0; i < 8; ++i) ps[i] += __shfl_xor(ps[i], msk, 32);
#pragma unroll
    for (int i = 0; i < 8; ++i) l[i] = l[i] * alpha[i] + ps[i];
#pragma unroll
    for (int c = 0; c < 16; ++c)
#pragma unroll
      for (int i = 0; i < 8; ++i) O[c][i] *= alpha[i];

    // ---- P: C-layout -> A-fragment via per-wave LDS buffer ----
#pragma unroll
    for (int i = 0; i < 8; ++i) {
      sP[wid][i + 8 * hi][ll]      = (__bf16)S0[i];
      sP[wid][i + 8 * hi][16 + ll] = (__bf16)S1[i];
    }
    const __bf16* pp = &sP[wid][ll][16 * hi];
    v16bf pf;
#pragma unroll
    for (int j = 0; j < 8; ++j) { pf[j] = pp[j]; pf[8 + j] = pp[8 + j]; }

    // ---- O += P * V  (V tile already transposed: contiguous B-frags) ----
#pragma unroll
    for (int c = 0; c < 16; ++c) {
      v16bf vf = *(const v16bf*)&sV[buf][c * 16 + ll][16 * hi];
      O[c] = wmma_bf16(pf, vf, O[c]);
    }
    __syncthreads();  // done with `buf` before wave0 prefetches over it
  }

  // ---- epilogue: out = O / l ----
  float* __restrict__ orow = out + ((size_t)b * LSEQ + q0) * HIDDEN;
#pragma unroll
  for (int c = 0; c < 16; ++c)
#pragma unroll
    for (int i = 0; i < 8; ++i) {
      const int row = i + 8 * hi;
      orow[(size_t)row * HIDDEN + c * 16 + ll] = O[c][i] / l[i];
    }
}

// ================= launch =================
extern "C" void kernel_launch(void* const* d_in, const int* in_sizes, int n_in,
                              void* d_out, int out_size, void* d_ws, size_t ws_size,
                              hipStream_t stream) {
  const float* X  = (const float*)d_in[0];
  const float* Wq = (const float*)d_in[1];
  const float* Wk = (const float*)d_in[2];
  const float* Wv = (const float*)d_in[3];
  // d_in[4] = lengths (unused by reference)
  float* out = (float*)d_out;

  const size_t elems = (size_t)BATCH * LSEQ * HIDDEN;  // 4M
  __bf16* Qb = (__bf16*)d_ws;           //  8 MB
  __bf16* Kb = Qb + elems;              //  8 MB
  __bf16* Vt = Kb + elems;              //  8 MB  (total ws use: 24 MB)

  dim3 g1((BATCH * LSEQ) / 128, HIDDEN / 64, 3);
  qkv_gemm_kernel<<<g1, 256, 0, stream>>>(X, Wq, Wk, Wv, Qb, Kb, Vt);

  dim3 g2(LSEQ / 64, BATCH, 1);
  attn_kernel<<<g2, 128, 0, stream>>>(Qb, Kb, Vt, out);
}
